// SelfAttention_54528904790729
// MI455X (gfx1250) — matrix operand order repaired and checked by reading, not compiled, not run on hardware
//
#include <hip/hip_runtime.h>
#include <hip/hip_bf16.h>
#include <math.h>

typedef __bf16 bf16;
typedef __bf16 v8bf  __attribute__((ext_vector_type(8)));
typedef __bf16 v16bf __attribute__((ext_vector_type(16)));
typedef float  v8f   __attribute__((ext_vector_type(8)));

#define B_ 32
#define N_ 1024
#define D_ 1024
#define H_ 1024

// Issue one 16B async global->LDS copy (CDNA5 GLOBAL_LOAD_ASYNC_TO_LDS_B128,
// tracked by ASYNCcnt). lds_off = byte offset within LDS (low 32 bits of a
// generic pointer to a __shared__ object), gaddr = 64-bit global address.
__device__ __forceinline__ void async_copy_b128(unsigned lds_off,
                                                const void* gaddr) {
  asm volatile("global_load_async_to_lds_b128 %0, %1, off"
               :
               : "v"(lds_off), "v"((unsigned long long)(uintptr_t)gaddr)
               : "memory");
}
__device__ __forceinline__ void async_wait0() {
  asm volatile("s_wait_asynccnt 0x0" ::: "memory");
}

// ---------------------------------------------------------------------------
// f32 -> bf16 conversion, 8 elements / thread (32B read, 16B write)
// ---------------------------------------------------------------------------
__global__ __launch_bounds__(256) void cvt_f32_bf16(const float* __restrict__ s,
                                                    bf16* __restrict__ d) {
  size_t i = ((size_t)blockIdx.x * 256 + threadIdx.x) * 8;
  float4 f0 = *(const float4*)(s + i);
  float4 f1 = *(const float4*)(s + i + 4);
  v8bf o;
  o[0] = (bf16)f0.x; o[1] = (bf16)f0.y; o[2] = (bf16)f0.z; o[3] = (bf16)f0.w;
  o[4] = (bf16)f1.x; o[5] = (bf16)f1.y; o[6] = (bf16)f1.z; o[7] = (bf16)f1.w;
  *(v8bf*)(d + i) = o;
}

// ---------------------------------------------------------------------------
// WMMA bf16 GEMM.  C[m][n] = sum_k A[m][k] * (TRANS_B ? Bm[n][k] : Bm[k][n])
// 128x128 block tile, BK=32, 8 waves, wave tile 32x64 (2x4 wmma accumulators).
// Double-buffered LDS; tile k+1 is staged with async global->LDS copies
// (ASYNCcnt) that overlap tile k's WMMAs. Non-transposed B (final GEMM) is
// transposed through registers instead. One barrier per K-step.
// All dims divide the tiles -> no guards, EXEC all-1s around WMMA.
// ---------------------------------------------------------------------------
template <bool TRANS_B, bool HAS_BIAS, bool OUT_F32, bool OUT_BF16>
__global__ __launch_bounds__(256) void gemm_wmma(
    const bf16* __restrict__ A, size_t strideA,
    const bf16* __restrict__ Bm, size_t strideB,
    const float* __restrict__ bias,
    float* __restrict__ Cf, size_t strideCf,
    bf16* __restrict__ Cb, size_t strideCb,
    int Ncols, int K) {
  constexpr int LDT = 40;  // 32 + 8 pad: 16B-aligned rows, conflict-free
  __shared__ bf16 As[2][128 * LDT];
  __shared__ bf16 Bs[2][128 * LDT];

  const int tid  = threadIdx.x;
  const int lane = tid & 31;
  const int wave = tid >> 5;
  const int lm   = lane & 15;   // lane's row/col within 16
  const int hi   = lane >> 4;   // K-group select (ISA 16-bit layout)
  const int wm0  = (wave >> 1) * 32;
  const int wn0  = (wave & 1) * 64;

  const int m0 = blockIdx.y * 128;
  const int n0 = blockIdx.x * 128;
  const size_t batch = blockIdx.z;

  const bf16* Ab = A + batch * strideA;
  const bf16* Bb = Bm + batch * strideB;

  const int rowA = tid >> 1;
  const int khA  = (tid & 1) * 16;

  // Per-thread global source pointers (advance by k each step)
  const bf16* gA = Ab + (size_t)(m0 + rowA) * K + khA;
  const bf16* gB = TRANS_B
      ? Bb + (size_t)(n0 + rowA) * K + khA
      : Bb + (size_t)(tid >> 3) * Ncols + n0 + (tid & 7) * 16;

  // LDS byte offsets of this thread's staging slots in each buffer
  const unsigned aLds[2] = {
      (unsigned)(uintptr_t)&As[0][rowA * LDT + khA],
      (unsigned)(uintptr_t)&As[1][rowA * LDT + khA]};
  const unsigned bLds[2] = {
      (unsigned)(uintptr_t)&Bs[0][rowA * LDT + khA],
      (unsigned)(uintptr_t)&Bs[1][rowA * LDT + khA]};

  v8f acc[2][4] = {};

  auto issue_async = [&](int b, int k0) {
    async_copy_b128(aLds[b], gA + k0);
    async_copy_b128(aLds[b] + 16, gA + k0 + 8);
    if (TRANS_B) {
      async_copy_b128(bLds[b], gB + k0);
      async_copy_b128(bLds[b] + 16, gB + k0 + 8);
    }
  };

  auto store_B_trans = [&](int b, v8bf b0, v8bf b1) {
    // transpose [K][Ncols] tile into k-contiguous LDS layout
    const int krow = tid >> 3;
    const int nn   = (tid & 7) * 16;
#pragma unroll
    for (int j = 0; j < 8; ++j) {
      Bs[b][(nn + j) * LDT + krow]     = b0[j];
      Bs[b][(nn + 8 + j) * LDT + krow] = b1[j];
    }
  };

  auto compute = [&](int b) {
    // A fragment (16x32): elems 0..7 -> K=hi*8+i, 8..15 -> K=16+hi*8+i
    v16bf afrag[2];
#pragma unroll
    for (int tm = 0; tm < 2; ++tm) {
      const bf16* p = &As[b][(wm0 + tm * 16 + lm) * LDT + hi * 8];
      v8bf lo  = *(const v8bf*)p;
      v8bf hi8 = *(const v8bf*)(p + 16);
      afrag[tm] = __builtin_shufflevector(lo, hi8, 0, 1, 2, 3, 4, 5, 6, 7, 8, 9,
                                          10, 11, 12, 13, 14, 15);
    }
    // B fragment (32x16): elems 0..7 -> K=hi*8+i, 8..15 -> K=16+hi*8+i
    v16bf bfrag[4];
#pragma unroll
    for (int tn = 0; tn < 4; ++tn) {
      const bf16* p = &Bs[b][(wn0 + tn * 16 + lm) * LDT + hi * 8];
      v8bf lo  = *(const v8bf*)p;
      v8bf hi8 = *(const v8bf*)(p + 16);
      bfrag[tn] = __builtin_shufflevector(lo, hi8, 0, 1, 2, 3, 4, 5, 6, 7, 8, 9,
                                          10, 11, 12, 13, 14, 15);
    }
#pragma unroll
    for (int tm = 0; tm < 2; ++tm)
#pragma unroll
      for (int tn = 0; tn < 4; ++tn)
        acc[tm][tn] = __builtin_amdgcn_wmma_f32_16x16x32_bf16(
            false, afrag[tm], false, bfrag[tn], (short)0, acc[tm][tn], false,
            false);
  };

  // --- prologue: tile 0 -> LDS[0] ---
  issue_async(0, 0);
  if (!TRANS_B) {
    v8bf b0 = *(const v8bf*)gB;
    v8bf b1 = *(const v8bf*)(gB + 8);
    store_B_trans(0, b0, b1);
  }
  async_wait0();
  __syncthreads();

  int buf = 0;
  for (int k0 = 32; k0 < K; k0 += 32) {
    // stage tile k0 into the other buffer; copies overlap the WMMAs below
    issue_async(buf ^ 1, k0);
    v8bf pb0, pb1;
    if (!TRANS_B) {
      const bf16* g = gB + (size_t)k0 * Ncols;
      pb0 = *(const v8bf*)g;
      pb1 = *(const v8bf*)(g + 8);
    }
    compute(buf);
    if (!TRANS_B) store_B_trans(buf ^ 1, pb0, pb1);
    async_wait0();
    __syncthreads();
    buf ^= 1;
  }
  compute(buf);

  // C/D layout: VGPR r -> row (r + 8*hi), col = lane&15
#pragma unroll
  for (int tm = 0; tm < 2; ++tm) {
#pragma unroll
    for (int tn = 0; tn < 4; ++tn) {
      int col = n0 + wn0 + tn * 16 + lm;
      float bv = 0.0f;
      if (HAS_BIAS) bv = bias[col];
#pragma unroll
      for (int r = 0; r < 8; ++r) {
        int row = m0 + wm0 + tm * 16 + hi * 8 + r;
        float v = acc[tm][tn][r] + bv;
        if (OUT_F32) Cf[batch * strideCf + (size_t)row * Ncols + col] = v;
        if (OUT_BF16) Cb[batch * strideCb + (size_t)row * Ncols + col] = (bf16)v;
      }
    }
  }
}

// ---------------------------------------------------------------------------
// Deterministic two-stage Frobenius-norm reduction (no float atomics)
// ---------------------------------------------------------------------------
__global__ __launch_bounds__(256) void sumsq_partial(const float* __restrict__ scores,
                                                     float* __restrict__ partials) {
  const int b = blockIdx.y;
  const float4* p = (const float4*)(scores + (size_t)b * N_ * N_ +
                                    (size_t)blockIdx.x * 65536);
  float s = 0.0f;
  for (int i = threadIdx.x; i < 16384; i += 256) {
    float4 v = p[i];
    s += v.x * v.x + v.y * v.y + v.z * v.z + v.w * v.w;
  }
  __shared__ float red[256];
  red[threadIdx.x] = s;
  __syncthreads();
  for (int off = 128; off > 0; off >>= 1) {
    if (threadIdx.x < off) red[threadIdx.x] += red[threadIdx.x + off];
    __syncthreads();
  }
  if (threadIdx.x == 0) partials[b * 16 + blockIdx.x] = red[0];
}

__global__ void sumsq_finish(const float* __restrict__ partials,
                             float* __restrict__ invnorms) {
  int b = threadIdx.x;
  if (b < B_) {
    float s = 0.0f;
    for (int i = 0; i < 16; ++i) s += partials[b * 16 + i];
    invnorms[b] = rsqrtf(s);
  }
}

// ---------------------------------------------------------------------------
// normalize -> tanh -> sigmoid gate -> transfer (in place) + bf16 copy
// 4 elements / thread: b128 in, b128 out, b64 bf16 out
// ---------------------------------------------------------------------------
__global__ __launch_bounds__(256) void gate_kernel(float* __restrict__ scores,
                                                   const float* __restrict__ gate_bias,
                                                   const float* __restrict__ invnorms,
                                                   bf16* __restrict__ trb) {
  size_t idx = ((size_t)blockIdx.x * 256 + threadIdx.x) * 4;
  int b  = (int)(idx >> 20);                // N*N = 1<<20
  int nm = (int)(idx & ((1u << 20) - 1));
  float inv = invnorms[b];
  float4 s  = *(const float4*)(scores + idx);
  float4 gb = *(const float4*)(gate_bias + nm);
  float t0 = tanhf(s.x * inv), t1 = tanhf(s.y * inv);
  float t2 = tanhf(s.z * inv), t3 = tanhf(s.w * inv);
  float4 tr;
  tr.x = t0 / (1.0f + __expf(-(fabsf(t0) + gb.x)));
  tr.y = t1 / (1.0f + __expf(-(fabsf(t1) + gb.y)));
  tr.z = t2 / (1.0f + __expf(-(fabsf(t2) + gb.z)));
  tr.w = t3 / (1.0f + __expf(-(fabsf(t3) + gb.w)));
  *(float4*)(scores + idx) = tr;
  v8bf o4;  // only 4 lanes used; store 8 bytes
  o4[0] = (bf16)tr.x; o4[1] = (bf16)tr.y; o4[2] = (bf16)tr.z; o4[3] = (bf16)tr.w;
  o4[4] = (bf16)0.f; o4[5] = (bf16)0.f; o4[6] = (bf16)0.f; o4[7] = (bf16)0.f;
  *(uint2*)(trb + idx) = *(uint2*)&o4;  // low 8 bytes = 4 bf16
}

// ---------------------------------------------------------------------------
extern "C" void kernel_launch(void* const* d_in, const int* in_sizes, int n_in,
                              void* d_out, int out_size, void* d_ws, size_t ws_size,
                              hipStream_t stream) {
  const float* x         = (const float*)d_in[0];
  const float* Wq        = (const float*)d_in[1];
  const float* bq        = (const float*)d_in[2];
  const float* Wk        = (const float*)d_in[3];
  const float* bk        = (const float*)d_in[4];
  const float* gate_bias = (const float*)d_in[5];

  float* out_next = (float*)d_out;                         // [B,N,D]
  float* out_tr   = (float*)d_out + (size_t)B_ * N_ * D_;  // [B,N,N] (also scratch)

  const size_t XB = (size_t)B_ * N_ * D_;  // x elems
  const size_t WB = (size_t)H_ * D_;       // weight elems
  const size_t QB = (size_t)B_ * N_ * H_;  // q/k elems
  const size_t SB = (size_t)B_ * N_ * N_;  // scores elems

  bf16* xb  = (bf16*)d_ws;
  bf16* wqb = xb + XB;
  bf16* wkb = wqb + WB;
  bf16* qb  = wkb + WB;
  bf16* kb  = qb + QB;
  bf16* trb = kb + QB;
  float* partials = (float*)(trb + SB);
  float* invnorms = partials + B_ * 16;

  // 1) f32 -> bf16 conversions
  cvt_f32_bf16<<<XB / (8 * 256), 256, 0, stream>>>(x, xb);
  cvt_f32_bf16<<<WB / (8 * 256), 256, 0, stream>>>(Wq, wqb);
  cvt_f32_bf16<<<WB / (8 * 256), 256, 0, stream>>>(Wk, wkb);

  // 2) q = x * Wq^T + bq ; k = x * Wk^T + bk   (M=B*N=32768, N=H, K=D)
  gemm_wmma<true, true, false, true><<<dim3(H_ / 128, (B_ * N_) / 128, 1), 256, 0, stream>>>(
      xb, 0, wqb, 0, bq, nullptr, 0, qb, 0, H_, D_);
  gemm_wmma<true, true, false, true><<<dim3(H_ / 128, (B_ * N_) / 128, 1), 256, 0, stream>>>(
      xb, 0, wkb, 0, bk, nullptr, 0, kb, 0, H_, D_);

  // 3) scores = q * k^T, batched (raw f32 into the transfer slot of d_out)
  gemm_wmma<true, false, true, false><<<dim3(N_ / 128, N_ / 128, B_), 256, 0, stream>>>(
      qb, (size_t)N_ * H_, kb, (size_t)N_ * H_, nullptr,
      out_tr, (size_t)N_ * N_, nullptr, 0, N_, H_);

  // 4) per-batch Frobenius norm (deterministic two-stage)
  sumsq_partial<<<dim3(16, B_), 256, 0, stream>>>(out_tr, partials);
  sumsq_finish<<<1, 32, 0, stream>>>(partials, invnorms);

  // 5) normalize + tanh + gate, in place; bf16 copy for last GEMM
  gate_kernel<<<SB / 1024, 256, 0, stream>>>(out_tr, gate_bias, invnorms, trb);

  // 6) next_states = transfer * x, batched (B operand row-major [K][Ncols])
  gemm_wmma<false, false, true, false><<<dim3(D_ / 128, N_ / 128, B_), 256, 0, stream>>>(
      trb, (size_t)N_ * N_, xb, (size_t)N_ * D_, nullptr,
      out_next, (size_t)N_ * D_, nullptr, 0, D_, N_);
}
